// FramePredictor_74844100100440
// MI455X (gfx1250) — matrix operand order repaired and checked by reading, not compiled, not run on hardware
//
#include <hip/hip_runtime.h>
#include <math.h>

// ---------------------------------------------------------------------------
// Model constants (match reference)
// ---------------------------------------------------------------------------
#define DMODEL 1024
#define NHEAD  8
#define HSZ    128
#define DFFN   4096
#define SEQ    1024
#define BATCH  8
#define BSREL  1024               // BS in reference (Er rows)
#define ROWS   (BATCH * SEQ)      // 8192 total token rows

typedef unsigned short u16;
typedef unsigned int   u32;
typedef __bf16 bf16;
typedef bf16  v16bf __attribute__((ext_vector_type(16)));
typedef u16   v16u  __attribute__((ext_vector_type(16)));
typedef float v8f   __attribute__((ext_vector_type(8)));
typedef u32   v4u   __attribute__((ext_vector_type(4)));
typedef float v4f   __attribute__((ext_vector_type(4)));

union VA { v16u u; v16bf b; v4u q[2]; };

__device__ __forceinline__ u16 f2b(float f) {
  // float -> bf16 round-to-nearest-even
  unsigned u = __float_as_uint(f);
  return (u16)((u + 0x7FFFu + ((u >> 16) & 1u)) >> 16);
}
__device__ __forceinline__ u32 pk2(float a, float b) {
  return (u32)f2b(a) | ((u32)f2b(b) << 16);
}
__device__ __forceinline__ float gelu_exact(float x) {
  return 0.5f * x * (1.0f + erff(x * 0.70710678118654752f));
}
__device__ __forceinline__ v8f vzero8() {
  v8f z;
#pragma unroll
  for (int i = 0; i < 8; ++i) z[i] = 0.f;
  return z;
}
// A-fragment (16x32 bf16): 8 halfs at p, 8 halfs at p+16  (two b128 loads)
__device__ __forceinline__ void ldfragA(VA& v, const u16* p) {
  v.q[0] = *(const v4u*)p;
  v.q[1] = *(const v4u*)(p + 16);
}
// B-fragment (32x16 bf16): 8 halfs at p, 8 halfs at p+16  (two b128 loads)
__device__ __forceinline__ void ldfragB(VA& v, const u16* p) {
  v.q[0] = *(const v4u*)p;
  v.q[1] = *(const v4u*)(p + 16);
}

// ---- CDNA5 async global->LDS copy (ASYNCcnt) --------------------------------
// Generic pointers to __shared__ data carry the LDS byte offset in addr[31:0]
// (ISA 10.2 aperture table), so truncation yields the VDST LDS address.
__device__ __forceinline__ u32 lds_off(const void* p) {
  return (u32)(size_t)p;
}
__device__ __forceinline__ void async_cp_b128(u32 ldsoff, const void* g) {
  asm volatile("global_load_async_to_lds_b128 %0, %1, off"
               :: "v"(ldsoff), "v"(g) : "memory");
}
__device__ __forceinline__ void wait_async0() {
  asm volatile("s_wait_asynccnt 0" ::: "memory");
}

enum { EPI_BF16 = 0, EPI_BF16_GELU = 1, EPI_F32_RES = 2, EPI_F32 = 3 };

// ---------------------------------------------------------------------------
// Generic bf16 WMMA GEMM:  C[M,N] = epi(A_bf16[M,K] @ W_f32[K,N] + bias)
// Tile: M=128 (8 waves x 16 rows), N=64 (4 WMMA n-tiles), K stepped by 32.
// A tile staged by the async LDS engine; W converted f32->bf16 through VGPRs
// (weights are L2-resident: a whole block's weights fit in the 192MB L2).
// ---------------------------------------------------------------------------
template <int MODE>
__global__ __launch_bounds__(256)
void gemm_bf16_wmma(const u16* __restrict__ A, int lda,
                    const float* __restrict__ W, int ldw,
                    const float* __restrict__ bias,
                    const float* __restrict__ R, int ldr,
                    void* __restrict__ C, int ldc,
                    int M, int N, int K) {
  __shared__ alignas(16) u16 As[128][48];   // A tile [m][k], 96B rows
  __shared__ alignas(16) u16 Bs[64][48];    // W tile transposed [n][k]

  const int tid  = threadIdx.x;
  const int wid  = tid >> 5;
  const int lane = tid & 31;
  const int lo   = lane & 15;
  const int hi   = lane >> 4;
  const int m0   = blockIdx.y * 128;
  const int n0   = blockIdx.x * 64;

  v8f acc[4];
#pragma unroll
  for (int t = 0; t < 4; ++t) acc[t] = vzero8();

  const int nkt = (K + 31) >> 5;
  for (int kt = 0; kt < nkt; ++kt) {
    const int k0 = kt << 5;
    const bool fullK = (k0 + 32) <= K;
    __syncthreads();
    // ---- stage A: 128x32, half a row (16 halfs = 32B) per thread ----
    {
      const int r  = tid >> 1;
      const int c0 = (tid & 1) * 16;
      if (fullK && (m0 + 128) <= M) {
        // async DMA: global -> LDS, no VGPR round-trip
        const u16* g = A + (size_t)(m0 + r) * lda + k0 + c0;
        const u32 d = lds_off(&As[r][c0]);
        async_cp_b128(d, g);
        async_cp_b128(d + 16, g + 8);
      } else {
        const bool rowok = (m0 + r) < M;
        const u16* a = A + (size_t)(rowok ? (m0 + r) : 0) * lda;
#pragma unroll
        for (int j = 0; j < 16; ++j) {
          const int k = k0 + c0 + j;
          const u16 v = a[(k < K) ? k : 0];
          As[r][c0 + j] = (rowok && k < K) ? v : (u16)0;
        }
      }
    }
    // ---- stage W transposed into Bs[n][k] (f32 -> bf16, packed u32) ----
    {
      const int kk = (tid & 15) * 2;          // even k within tile
      const int nb = (tid >> 4) * 4;          // 4 consecutive n per thread
      if (fullK && (n0 + 64) <= N && ((ldw & 3) == 0)) {
        const v4f w0 = *(const v4f*)&W[(size_t)(k0 + kk) * ldw + n0 + nb];
        const v4f w1 = *(const v4f*)&W[(size_t)(k0 + kk + 1) * ldw + n0 + nb];
        *(u32*)&Bs[nb + 0][kk] = pk2(w0.x, w1.x);
        *(u32*)&Bs[nb + 1][kk] = pk2(w0.y, w1.y);
        *(u32*)&Bs[nb + 2][kk] = pk2(w0.z, w1.z);
        *(u32*)&Bs[nb + 3][kk] = pk2(w0.w, w1.w);
      } else {
        const bool k0ok = (k0 + kk) < K;
        const bool k1ok = (k0 + kk + 1) < K;
        const size_t r0 = (size_t)(k0ok ? (k0 + kk) : 0) * ldw;
        const size_t r1 = (size_t)(k1ok ? (k0 + kk + 1) : 0) * ldw;
#pragma unroll
        for (int j = 0; j < 4; ++j) {
          const int n = n0 + nb + j;
          const int nc = (n < N) ? n : 0;
          const float a0 = (k0ok && n < N) ? W[r0 + nc] : 0.f;
          const float a1 = (k1ok && n < N) ? W[r1 + nc] : 0.f;
          *(u32*)&Bs[nb + j][kk] = pk2(a0, a1);
        }
      }
    }
    // prefetch next k-tile while WMMAs run (global_prefetch_b8)
    if (kt + 1 < nkt) {
      __builtin_prefetch(A + (size_t)(m0 + (tid >> 1)) * lda + k0 + 32, 0, 3);
      __builtin_prefetch(&W[(size_t)(k0 + 32 + (tid & 31)) * ldw + n0], 0, 3);
    }
    wait_async0();
    __syncthreads();

    VA af;
    ldfragA(af, &As[wid * 16 + lo][hi * 8]);
#pragma unroll
    for (int t = 0; t < 4; ++t) {
      VA bf;
      ldfragB(bf, &Bs[t * 16 + lo][hi * 8]);
      acc[t] = __builtin_amdgcn_wmma_f32_16x16x32_bf16(
          false, af.b, false, bf.b, (short)0, acc[t], false, false);
    }
  }

  // ---- epilogue (C layout: row = r + 8*hi, col = lane&15 within n-tile) ----
#pragma unroll
  for (int t = 0; t < 4; ++t) {
    const int col = n0 + t * 16 + lo;
    const float bs = (col < N) ? bias[col] : 0.f;
#pragma unroll
    for (int r = 0; r < 8; ++r) {
      const int row = m0 + wid * 16 + r + 8 * hi;
      if (row < M && col < N) {
        float v = acc[t][r] + bs;
        if (MODE == EPI_BF16_GELU) v = gelu_exact(v);
        if (MODE == EPI_F32_RES)   v += R[(size_t)row * ldr + col];
        if (MODE == EPI_BF16 || MODE == EPI_BF16_GELU)
          ((u16*)C)[(size_t)row * ldc + col] = f2b(v);
        else
          ((float*)C)[(size_t)row * ldc + col] = v;
      }
    }
  }
}

// ---------------------------------------------------------------------------
// LayerNorm (f32 in -> bf16 out), one row per workgroup
// ---------------------------------------------------------------------------
__global__ __launch_bounds__(256)
void layernorm_rows(const float* __restrict__ X,
                    const float* __restrict__ g,
                    const float* __restrict__ bta,
                    u16* __restrict__ Y, int D) {
  __shared__ float red[256];
  const int row = blockIdx.x;
  const int tid = threadIdx.x;
  const float* x = X + (size_t)row * D;
  float s = 0.f, s2 = 0.f;
  for (int i = tid; i < D; i += 256) { float v = x[i]; s += v; s2 += v * v; }
  red[tid] = s;  __syncthreads();
  for (int off = 128; off > 0; off >>= 1) { if (tid < off) red[tid] += red[tid + off]; __syncthreads(); }
  const float mu = red[0] / (float)D;
  __syncthreads();
  red[tid] = s2; __syncthreads();
  for (int off = 128; off > 0; off >>= 1) { if (tid < off) red[tid] += red[tid + off]; __syncthreads(); }
  const float var = red[0] / (float)D - mu * mu;
  const float inv = rsqrtf(var + 1e-5f);
  for (int i = tid; i < D; i += 256)
    Y[(size_t)row * D + i] = f2b((x[i] - mu) * inv * g[i] + bta[i]);
}

// LayerNorm over virtual concatenation [X(1024) | C1(w1) | C2(w2) | C3(w3)]
// -> bf16 out with stride ldy, zero-padded beyond din.
__global__ __launch_bounds__(256)
void layernorm_head(const float* __restrict__ X,
                    const float* __restrict__ C1, int w1,
                    const float* __restrict__ C2, int w2,
                    const float* __restrict__ C3, int w3,
                    const float* __restrict__ g,
                    const float* __restrict__ bta,
                    u16* __restrict__ Y, int ldy) {
  __shared__ float red[256];
  const int row = blockIdx.x;
  const int tid = threadIdx.x;
  const int din = DMODEL + w1 + w2 + w3;

  auto fetch = [&](int i) -> float {
    if (i < DMODEL) return X[(size_t)row * DMODEL + i];
    i -= DMODEL;
    if (i < w1) return C1[(size_t)row * w1 + i];
    i -= w1;
    if (i < w2) return C2[(size_t)row * w2 + i];
    i -= w2;
    return C3[(size_t)row * w3 + i];
  };

  float s = 0.f, s2 = 0.f;
  for (int i = tid; i < din; i += 256) { float v = fetch(i); s += v; s2 += v * v; }
  red[tid] = s;  __syncthreads();
  for (int off = 128; off > 0; off >>= 1) { if (tid < off) red[tid] += red[tid + off]; __syncthreads(); }
  const float mu = red[0] / (float)din;
  __syncthreads();
  red[tid] = s2; __syncthreads();
  for (int off = 128; off > 0; off >>= 1) { if (tid < off) red[tid] += red[tid + off]; __syncthreads(); }
  const float var = red[0] / (float)din - mu * mu;
  const float inv = rsqrtf(var + 1e-5f);
  for (int i = tid; i < ldy; i += 256) {
    u16 o = 0;
    if (i < din) o = f2b((fetch(i) - mu) * inv * g[i] + bta[i]);
    Y[(size_t)row * ldy + i] = o;
  }
}

// ---------------------------------------------------------------------------
// Flash attention with relative-position band bias.
//   scores[i,j] = (q_i.k_j + q_i.Er[BS-1-(i-j)]) / sqrt(HS),  causal mask
// One workgroup = (batch, head, 64 query rows); 4 waves, 16 q rows each.
// Band trick: stepping the j-tile by 16 shifts the Er band by 16, so the
// previous "high" band accumulator becomes the new "low" band.
// ---------------------------------------------------------------------------
__global__ __launch_bounds__(128)
void attn_relpos_wmma(const u16* __restrict__ QKV,   // [rows][3072] bf16
                      const float* __restrict__ Er,  // [BSREL][HSZ] f32
                      u16* __restrict__ O) {         // [rows][DMODEL] bf16
  __shared__ alignas(16) u16  Kt[64][HSZ + 8];       // K tile [j][d], 272B rows
  __shared__ alignas(16) u16  Vt[HSZ][64 + 8];       // V tile transposed [d][j]
  __shared__ alignas(16) u16  ErS[4][16][HSZ + 8];   // per-wave Er band [t'][d]
  __shared__ alignas(16) u16  Ps[4][16][48];         // per-wave probs 16x32
  __shared__ alignas(16) float Ds[4][16][34];        // per-wave band 16x32 f32

  const int qblk = blockIdx.x;           // 0..15
  const int h    = blockIdx.y;           // 0..7
  const int b    = blockIdx.z;           // 0..7
  const int tid  = threadIdx.x;
  const int wid  = tid >> 5;
  const int lane = tid & 31;
  const int lo   = lane & 15;
  const int hi   = lane >> 4;
  const int i0w  = qblk * 64 + wid * 16;           // wave's first query row
  const size_t rowbase = (size_t)b * SEQ;
  const float scale = 0.08838834764831845f;        // 1/sqrt(128)

  // Q fragments for this wave's 16 rows, K = 4 chunks of 32 (b128 loads)
  VA qf[4];
  {
    const size_t qrow = (rowbase + i0w + lo) * 3072 + (size_t)h * HSZ;
#pragma unroll
    for (int c = 0; c < 4; ++c) {
      const u16* qp = QKV + qrow + c * 32 + hi * 8;
      qf[c].q[0] = *(const v4u*)qp;
      qf[c].q[1] = *(const v4u*)(qp + 16);
    }
  }

  // band(ebase): D[di,t'] = q_(i0w+di) . Er[ebase+t'],  t' in 0..15
  auto band = [&](int ebase) -> v8f {
    {
      const int t  = lo;
      const int dh = hi * 64;
      const int e  = ebase + t;
      const bool ok = ((unsigned)e < (unsigned)BSREL);
      const u32 msk = ok ? 0xFFFFFFFFu : 0u;
      const v4f* er4 = (const v4f*)(Er + (size_t)(ok ? e : 0) * HSZ + dh);
      v4u* dst = (v4u*)&ErS[wid][t][dh];
#pragma unroll
      for (int c8 = 0; c8 < 8; ++c8) {
        const v4f f0 = er4[2 * c8];
        const v4f f1 = er4[2 * c8 + 1];
        v4u o;
        o.x = pk2(f0.x, f0.y) & msk;
        o.y = pk2(f0.z, f0.w) & msk;
        o.z = pk2(f1.x, f1.y) & msk;
        o.w = pk2(f1.z, f1.w) & msk;
        dst[c8] = o;
      }
    }
    asm volatile("s_wait_dscnt 0" ::: "memory");   // wave-local LDS RAW
    v8f d = vzero8();
#pragma unroll
    for (int c = 0; c < 4; ++c) {
      VA bf;
      ldfragB(bf, &ErS[wid][lo][c * 32 + hi * 8]);
      d = __builtin_amdgcn_wmma_f32_16x16x32_bf16(
          false, qf[c].b, false, bf.b, (short)0, d, false, false);
    }
    return d;
  };

  float m[8], ssum[8];
  v8f yacc[8];
#pragma unroll
  for (int r = 0; r < 8; ++r) { m[r] = -3.0e38f; ssum[r] = 0.f; }
#pragma unroll
  for (int t = 0; t < 8; ++t) yacc[t] = vzero8();

  // seed: "high" band of virtual previous step == low band of js=0
  v8f Dhi = band(BSREL - 16 - i0w);

  const int nchunk = qblk + 1;             // j covers [0, qblk*64+63]
  for (int ch = 0; ch < nchunk; ++ch) {
    const int j0 = ch * 64;
    __syncthreads();
    // cooperative stage: K via async LDS engine, V transposed through VGPRs
    {
      const int j  = tid >> 1;
      const int dh = (tid & 1) * 64;
      const size_t kr = (rowbase + j0 + j) * 3072 + 1024 + (size_t)h * HSZ + dh;
      const size_t vr = (rowbase + j0 + j) * 3072 + 2048 + (size_t)h * HSZ + dh;
      const u16* kg = QKV + kr;
      const u32 kd = lds_off(&Kt[j][dh]);
#pragma unroll
      for (int t = 0; t < 8; ++t) async_cp_b128(kd + t * 16, kg + t * 8);
      v4u vv[8];
      const v4u* vsrc = (const v4u*)(QKV + vr);
#pragma unroll
      for (int t = 0; t < 8; ++t) vv[t] = vsrc[t];
      const u16* vs = (const u16*)vv;
#pragma unroll
      for (int t = 0; t < 64; ++t) Vt[dh + t][j] = vs[t];
    }
    wait_async0();
    __syncthreads();

    for (int sub = 0; sub < 4; ++sub) {
      const int js = j0 + sub * 16;
      // shift band and compute new high half: e = BSREL - i0w + js + t'
      v8f Dlo = Dhi;
      Dhi = band(BSREL - i0w + js);
#pragma unroll
      for (int r = 0; r < 8; ++r) {
        const int di = r + 8 * hi;
        Ds[wid][di][lo]      = Dlo[r];
        Ds[wid][di][16 + lo] = Dhi[r];
      }

      // scores tile 16x16: Q @ K^T over K=128
      v8f s = vzero8();
#pragma unroll
      for (int c = 0; c < 4; ++c) {
        VA kf;
        ldfragB(kf, &Kt[sub * 16 + lo][c * 32 + hi * 8]);
        s = __builtin_amdgcn_wmma_f32_16x16x32_bf16(
            false, qf[c].b, false, kf.b, (short)0, s, false, false);
      }
      asm volatile("s_wait_dscnt 0" ::: "memory");

      // online softmax update (a row lives across a 16-lane half)
#pragma unroll
      for (int r = 0; r < 8; ++r) {
        const int di = r + 8 * hi;
        const int i  = i0w + di;
        const int j  = js + lo;
        float val = (s[r] + Ds[wid][di][15 - di + lo]) * scale;
        if (j > i) val = -3.0e38f;
        float rv = val;
#pragma unroll
        for (int off = 1; off < 16; off <<= 1) rv = fmaxf(rv, __shfl_xor(rv, off));
        const float mn   = fmaxf(m[r], rv);
        const float corr = expf(m[r] - mn);
        const float pv   = expf(val - mn);
        float pr = pv;
#pragma unroll
        for (int off = 1; off < 16; off <<= 1) pr += __shfl_xor(pr, off);
        ssum[r] = ssum[r] * corr + pr;
        m[r] = mn;
#pragma unroll
        for (int t = 0; t < 8; ++t) yacc[t][r] *= corr;
        Ps[wid][di][(sub & 1) * 16 + lo] = f2b(pv);
      }

      if (sub & 1) {                       // PV over K=32 (two 16-j subtiles)
        asm volatile("s_wait_dscnt 0" ::: "memory");
        VA pf;
        ldfragA(pf, &Ps[wid][lo][hi * 8]);
        const int pairbase = (sub >> 1) << 5;
#pragma unroll
        for (int t = 0; t < 8; ++t) {
          VA vf;
          ldfragB(vf, &Vt[t * 16 + lo][pairbase + hi * 8]);
          yacc[t] = __builtin_amdgcn_wmma_f32_16x16x32_bf16(
              false, pf.b, false, vf.b, (short)0, yacc[t], false, false);
        }
      }
    }
  }

  // normalize and store (bf16) into O[row][h*128 + d]
#pragma unroll
  for (int t = 0; t < 8; ++t) {
#pragma unroll
    for (int r = 0; r < 8; ++r) {
      const int di = r + 8 * hi;
      const size_t row = rowbase + i0w + di;
      O[row * DMODEL + (size_t)h * HSZ + t * 16 + lo] = f2b(yacc[t][r] / ssum[r]);
    }
  }
}

// ---------------------------------------------------------------------------
// Host-side orchestration
// ---------------------------------------------------------------------------
static void launch_gemm(int mode, dim3 grid, hipStream_t stream,
                        const u16* A, int lda, const float* W, int ldw,
                        const float* bias, const float* R, int ldr,
                        void* C, int ldc, int M, int N, int K) {
  const dim3 blk(256);
  switch (mode) {
    case EPI_BF16:
      gemm_bf16_wmma<EPI_BF16><<<grid, blk, 0, stream>>>(A, lda, W, ldw, bias, R, ldr, C, ldc, M, N, K);
      break;
    case EPI_BF16_GELU:
      gemm_bf16_wmma<EPI_BF16_GELU><<<grid, blk, 0, stream>>>(A, lda, W, ldw, bias, R, ldr, C, ldc, M, N, K);
      break;
    case EPI_F32_RES:
      gemm_bf16_wmma<EPI_F32_RES><<<grid, blk, 0, stream>>>(A, lda, W, ldw, bias, R, ldr, C, ldc, M, N, K);
      break;
    default:
      gemm_bf16_wmma<EPI_F32><<<grid, blk, 0, stream>>>(A, lda, W, ldw, bias, R, ldr, C, ldc, M, N, K);
      break;
  }
}

extern "C" void kernel_launch(void* const* d_in, const int* in_sizes, int n_in,
                              void* d_out, int out_size, void* d_ws, size_t ws_size,
                              hipStream_t stream) {
  (void)in_sizes; (void)n_in; (void)out_size; (void)ws_size;
  auto f = [&](int i) { return (const float*)d_in[i]; };

  // workspace partition (~186 MB total)
  char* w = (char*)d_ws;
  float* Xa = (float*)w;   w += (size_t)ROWS * DMODEL * 4;   // 32 MB
  float* Xb = (float*)w;   w += (size_t)ROWS * DMODEL * 4;   // 32 MB
  u16*   LN = (u16*)w;     w += (size_t)ROWS * DMODEL * 2;   // 16 MB
  u16*   BIG = (u16*)w;    w += (size_t)ROWS * DFFN * 2;     // 64 MB (QKV / FFN hidden / head hidden)
  u16*   ATTN = (u16*)w;   w += (size_t)ROWS * DMODEL * 2;   // 16 MB
  u16*   HEADIN = (u16*)w; w += (size_t)ROWS * 1120 * 2;     // 18 MB (padded concat-LN)

  const dim3 blk256(256), blk128(128);
  const dim3 gLN(ROWS);
  const dim3 gAttn(SEQ / 64, NHEAD, BATCH);

  const float* cur = f(0);   // block 0 reads the (const) input x
  for (int blkI = 0; blkI < 4; ++blkI) {
    const int pb = 1 + blkI * 13;
    // ln1 -> bf16
    layernorm_rows<<<gLN, blk256, 0, stream>>>(cur, f(pb + 0), f(pb + 1), LN, DMODEL);
    // qkv = ln @ Wqkv + b  (bf16 out)
    launch_gemm(EPI_BF16, dim3(3 * DMODEL / 64, ROWS / 128), stream,
                LN, DMODEL, f(pb + 2), 3 * DMODEL, f(pb + 3), nullptr, 0,
                BIG, 3 * DMODEL, ROWS, 3 * DMODEL, DMODEL);
    // attention with relative position bias
    attn_relpos_wmma<<<gAttn, blk128, 0, stream>>>(BIG, f(pb + 6), ATTN);
    // x = x + attn @ Wproj + b  (f32 out with residual)
    launch_gemm(EPI_F32_RES, dim3(DMODEL / 64, ROWS / 128), stream,
                ATTN, DMODEL, f(pb + 4), DMODEL, f(pb + 5), cur, DMODEL,
                Xb, DMODEL, ROWS, DMODEL, DMODEL);
    // ln2 -> bf16
    layernorm_rows<<<gLN, blk256, 0, stream>>>(Xb, f(pb + 7), f(pb + 8), LN, DMODEL);
    // h = gelu(ln @ Wfc + b)  (bf16 out)
    launch_gemm(EPI_BF16_GELU, dim3(DFFN / 64, ROWS / 128), stream,
                LN, DMODEL, f(pb + 9), DFFN, f(pb + 10), nullptr, 0,
                BIG, DFFN, ROWS, DFFN, DMODEL);
    // x = x + h @ Wfp + b  (f32 out with residual)
    launch_gemm(EPI_F32_RES, dim3(DMODEL / 64, ROWS / 128), stream,
                BIG, DFFN, f(pb + 11), DMODEL, f(pb + 12), Xb, DMODEL,
                Xa, DMODEL, ROWS, DMODEL, DFFN);
    cur = Xa;
  }

  // output slices: (main, shoulder, cdir, btn) concatenated flat
  float* out    = (float*)d_out;
  float* o_main = out;
  float* o_sh   = out + (size_t)ROWS * 63;
  float* o_cd   = o_sh + (size_t)ROWS * 3;
  float* o_bt   = o_cd + (size_t)ROWS * 5;
  u16* HHID = BIG;   // reuse big buffer for head hidden (<= 547 cols, ld 560)

  auto run_head = [&](int hb,
                      const float* c1, int w1, const float* c2, int w2,
                      const float* c3, int w3, int hdim, int dout, float* op) {
    const int din = DMODEL + w1 + w2 + w3;
    layernorm_head<<<gLN, blk256, 0, stream>>>(
        Xa, c1, w1, c2, w2, c3, w3, f(hb + 0), f(hb + 1), HEADIN, 1120);
    launch_gemm(EPI_BF16_GELU, dim3((hdim + 63) / 64, ROWS / 128), stream,
                HEADIN, 1120, f(hb + 2), hdim, f(hb + 3), nullptr, 0,
                HHID, 560, ROWS, hdim, din);
    launch_gemm(EPI_F32, dim3((dout + 63) / 64, ROWS / 128), stream,
                HHID, 560, f(hb + 4), dout, f(hb + 5), nullptr, 0,
                op, dout, ROWS, dout, hdim);
  };

  run_head(53, nullptr, 0, nullptr, 0, nullptr, 0, 512, 3, o_sh);
  run_head(59, o_sh, 3, nullptr, 0, nullptr, 0,   513, 5, o_cd);
  run_head(65, o_sh, 3, o_cd, 5, nullptr, 0,      516, 63, o_main);
  run_head(71, o_sh, 3, o_cd, 5, o_main, 63,      547, 5, o_bt);
}
